// DynamicSemanticModule_79637283602497
// MI455X (gfx1250) — matrix operand order repaired and checked by reading, not compiled, not run on hardware
//
#include <hip/hip_runtime.h>

typedef __attribute__((ext_vector_type(16))) _Float16 v16h;
typedef __attribute__((ext_vector_type(8)))  float    v8f;
typedef __attribute__((ext_vector_type(8)))  unsigned v8u;
typedef __attribute__((ext_vector_type(4)))  unsigned v4u;

#define B_    32
#define LT    1024
#define LP    512
#define EMO_  1024
#define FEAT_ 1024

static __device__ __forceinline__ float wave_max(float v) {
#pragma unroll
  for (int o = 16; o; o >>= 1) v = fmaxf(v, __shfl_xor(v, o, 32));
  return v;
}
static __device__ __forceinline__ float wave_sum(float v) {
#pragma unroll
  for (int o = 16; o; o >>= 1) v += __shfl_xor(v, o, 32);
  return v;
}

// Async DMA of 16 bytes global -> LDS (gfx1250, ASYNCcnt-tracked; no VGPR round trip)
static __device__ __forceinline__ void async_copy_b128(void* lds_ptr,
                                                       const _Float16* gptr) {
  unsigned lds = (unsigned)(unsigned long long)lds_ptr;  // low 32 bits = LDS offset
  asm volatile("global_load_async_to_lds_b128 %0, %1, off"
               :: "v"(lds), "v"(gptr) : "memory");
}
static __device__ __forceinline__ void wait_async0() {
  asm volatile("s_wait_asynccnt 0x0" ::: "memory");
}

// ---------------------------------------------------------------------------
// Generic batched WMMA GEMM: C[z] = alpha * A[z] @ B[z] + bias
// Block tile 64x64, K-step 32, 4 waves (wave32), one 16-row strip per wave.
// Double-buffered LDS: async-DMA prefetch of tile t+1 overlaps WMMA on tile t.
// A: fp16 row-major [M,Kd]. B: fp16; bT=1 -> [N,Kd] row-major (async-DMA staged
// N-major at zero cost; all weights are pre-transposed to hit this path),
// bT=0 -> [Kd,N] row-major (coalesced VALU transpose gather; activations only).
// LDS rows padded to 40 halves (80B): 16B-aligned b128 fragment loads with a
// 20-bank skew that tiles all 64 banks conflict-free.
// ---------------------------------------------------------------------------
__global__ __launch_bounds__(128)
void wmma_gemm_kernel(const _Float16* __restrict__ A, long sA, int lda,
                      const _Float16* __restrict__ Bm, long sB, int ldb, int bT,
                      float* __restrict__ C32, long sC32,
                      _Float16* __restrict__ C16, long sC16, int ldc,
                      const float* __restrict__ bias, float alpha, int Kd) {
  __shared__ _Float16 As[2][64][40];
  __shared__ _Float16 Bs[2][64][40];   // N-major: Bs[buf][n][k]
  const int tid = threadIdx.x, wave = tid >> 5, lane = tid & 31;
  const int mBlk = blockIdx.x * 64, nBlk = blockIdx.y * 64;
  const _Float16* Ab = A + (long)blockIdx.z * sA;
  const _Float16* Bb = Bm + (long)blockIdx.z * sB;
  v8f acc0 = {}, acc1 = {}, acc2 = {}, acc3 = {};

  auto stage = [&](int buf, int k0) {
#pragma unroll
    for (int j = 0; j < 2; ++j) {        // A tile 64x32 halves = 256 x 16B DMA
      const int idx = tid + 128 * j;
      const int r = idx >> 2, ch = (idx & 3) << 3;
      async_copy_b128(&As[buf][r][ch], Ab + (long)(mBlk + r) * lda + k0 + ch);
    }
    if (bT) {
#pragma unroll
      for (int j = 0; j < 2; ++j) {
        const int idx = tid + 128 * j;
        const int n = idx >> 2, ch = (idx & 3) << 3;
        async_copy_b128(&Bs[buf][n][ch], Bb + (long)(nBlk + n) * ldb + k0 + ch);
      }
    } else {
      // transpose gather, coalesced along N; pack half-pairs as u32 stores
#pragma unroll
      for (int j = 0; j < 8; ++j) {
        const int idx = tid + 128 * j;   // 1024 half-pairs
        const int n = idx & 63, c = (idx >> 6) << 1;
        const unsigned short lo = __builtin_bit_cast(
            unsigned short, Bb[(long)(k0 + c) * ldb + nBlk + n]);
        const unsigned short hi = __builtin_bit_cast(
            unsigned short, Bb[(long)(k0 + c + 1) * ldb + nBlk + n]);
        *(unsigned*)&Bs[buf][n][c] = (unsigned)lo | ((unsigned)hi << 16);
      }
    }
  };

  const int mrow = (wave << 4) + (lane & 15);
  const int akb = (lane >> 4) << 3;
  const int bkb = (lane >> 4) << 3;
  const int nl = lane & 15;

  auto loadB = [&](int buf, int n) -> v16h {
    const v4u b0 = *(const v4u*)&Bs[buf][n][bkb];
    const v4u b1 = *(const v4u*)&Bs[buf][n][bkb + 16];
    v8u u;
    u[0] = b0[0]; u[1] = b0[1]; u[2] = b0[2]; u[3] = b0[3];
    u[4] = b1[0]; u[5] = b1[1]; u[6] = b1[2]; u[7] = b1[3];
    return __builtin_bit_cast(v16h, u);
  };

  stage(0, 0);
  int cur = 0;
  for (int k0 = 0; k0 < Kd; k0 += 32) {
    wait_async0();
    __syncthreads();                       // tile[cur] staged by all waves
    if (k0 + 32 < Kd) stage(cur ^ 1, k0 + 32);  // prefetch overlaps WMMA below

    // A fragment (16x32 f16): lanes 0-15 M=0..15 K{0..7,16..23}; lanes 16-31 K{8..15,24..31}
    const v4u a0 = *(const v4u*)&As[cur][mrow][akb];
    const v4u a1 = *(const v4u*)&As[cur][mrow][akb + 16];
    v8u af;
    af[0] = a0[0]; af[1] = a0[1]; af[2] = a0[2]; af[3] = a0[3];
    af[4] = a1[0]; af[5] = a1[1]; af[6] = a1[2]; af[7] = a1[3];
    const v16h a = __builtin_bit_cast(v16h, af);
    // B fragments (32x16 f16): lanes 0-15 K{0..7,16..23}, lanes 16-31 K{8..15,24..31}, N=lane%16
    const v16h bf0 = loadB(cur, nl);
    const v16h bf1 = loadB(cur, 16 + nl);
    const v16h bf2 = loadB(cur, 32 + nl);
    const v16h bf3 = loadB(cur, 48 + nl);
    acc0 = __builtin_amdgcn_wmma_f32_16x16x32_f16(false, a, false, bf0, (short)0, acc0, false, false);
    acc1 = __builtin_amdgcn_wmma_f32_16x16x32_f16(false, a, false, bf1, (short)0, acc1, false, false);
    acc2 = __builtin_amdgcn_wmma_f32_16x16x32_f16(false, a, false, bf2, (short)0, acc2, false, false);
    acc3 = __builtin_amdgcn_wmma_f32_16x16x32_f16(false, a, false, bf3, (short)0, acc3, false, false);

    __syncthreads();                       // all reads of tile[cur] done
    cur ^= 1;
  }

  // epilogue: C/D layout lanes 0-15 N=lane M=r, lanes 16-31 N=lane-16 M=8+r
  const int mbase = mBlk + (wave << 4) + ((lane >> 4) << 3);
  float* Cb32 = C32 ? C32 + (long)blockIdx.z * sC32 : nullptr;
  _Float16* Cb16 = C16 ? C16 + (long)blockIdx.z * sC16 : nullptr;
#pragma unroll
  for (int nt = 0; nt < 4; ++nt) {
    const v8f acc = nt == 0 ? acc0 : nt == 1 ? acc1 : nt == 2 ? acc2 : acc3;
    const int n = nBlk + (nt << 4) + nl;
    const float bv = bias ? bias[n] : 0.0f;
#pragma unroll
    for (int r = 0; r < 8; ++r) {
      const float v = acc[r] * alpha + bv;
      const long off = (long)(mbase + r) * ldc + n;
      if (Cb32) Cb32[off] = v;
      if (Cb16) Cb16[off] = (_Float16)v;
    }
  }
}

// ---------------------------------------------------------------------------
// Elementwise / reduction helper kernels
// ---------------------------------------------------------------------------
// one-time: convert weight [K,N] f32 -> TRANSPOSED [N,K] f16 (async-DMA B path)
__global__ void transpose_w_kernel(const float* __restrict__ in,
                                   _Float16* __restrict__ out, int K, int N) {
  int i = blockIdx.x * 256 + threadIdx.x;   // over N*K, contiguous in k
  if (i >= K * N) return;
  int n = i / K, k = i - n * K;
  out[i] = (_Float16)in[(long)k * N + n];
}

// AdaptiveAvgPool1d(256) over Lp=512 == exact pair mean; emit fp16
__global__ void pool_phr_kernel(const float* __restrict__ phr,
                                _Float16* __restrict__ out) {
  int i = blockIdx.x * 256 + threadIdx.x;   // B*256*1024
  int f = i & 1023, v = (i >> 10) & 255, b = i >> 18;
  long base = (long)b * LP * FEAT_ + (long)(2 * v) * FEAT_ + f;
  out[i] = (_Float16)(0.5f * (phr[base] + phr[base + FEAT_]));
}

__global__ void video_to_combine_kernel(const float* __restrict__ video,
                                        _Float16* __restrict__ combine) {
  int i = blockIdx.x * 256 + threadIdx.x;   // B*256*1024
  int f = i & 1023, v = (i >> 10) & 255, b = i >> 18;
  combine[(long)b * 512 * 1024 + (long)(256 + v) * 1024 + f] = (_Float16)video[i];
}

// softmax over 128 cols, write TRANSPOSED fp16 [B,128,512] (A operand of agg GEMM)
__global__ __launch_bounds__(128)
void softmax128_T_kernel(const float* __restrict__ logits,
                         _Float16* __restrict__ outT) {
  int row = blockIdx.x * 4 + (threadIdx.x >> 5);   // B*512
  int lane = threadIdx.x & 31;
  int b = row >> 9, m = row & 511;
  const float* p = logits + (long)row * 128;
  float x0 = p[lane], x1 = p[lane + 32], x2 = p[lane + 64], x3 = p[lane + 96];
  float mx = wave_max(fmaxf(fmaxf(x0, x1), fmaxf(x2, x3)));
  float e0 = __expf(x0 - mx), e1 = __expf(x1 - mx), e2 = __expf(x2 - mx), e3 = __expf(x3 - mx);
  float inv = 1.0f / wave_sum(e0 + e1 + e2 + e3);
  _Float16* o = outT + (long)b * 128 * 512 + m;
  o[(long)lane * 512]        = (_Float16)(e0 * inv);
  o[(long)(lane + 32) * 512] = (_Float16)(e1 * inv);
  o[(long)(lane + 64) * 512] = (_Float16)(e2 * inv);
  o[(long)(lane + 96) * 512] = (_Float16)(e3 * inv);
}

// softmax over 128 cols, row-major fp16 out (attention probs P)
__global__ __launch_bounds__(128)
void softmax128_kernel(const float* __restrict__ S, _Float16* __restrict__ P) {
  long row = blockIdx.x * 4 + (threadIdx.x >> 5);  // B*3*1024
  int lane = threadIdx.x & 31;
  const float* p = S + row * 128;
  float x0 = p[lane], x1 = p[lane + 32], x2 = p[lane + 64], x3 = p[lane + 96];
  float mx = wave_max(fmaxf(fmaxf(x0, x1), fmaxf(x2, x3)));
  float e0 = __expf(x0 - mx), e1 = __expf(x1 - mx), e2 = __expf(x2 - mx), e3 = __expf(x3 - mx);
  float inv = 1.0f / wave_sum(e0 + e1 + e2 + e3);
  _Float16* o = P + row * 128;
  o[lane] = (_Float16)(e0 * inv);      o[lane + 32] = (_Float16)(e1 * inv);
  o[lane + 64] = (_Float16)(e2 * inv); o[lane + 96] = (_Float16)(e3 * inv);
}

// mean over Lt of emo -> [B,1024]  (A_pool = emo_mean @ agg^T : pooled first)
__global__ void emo_mean_kernel(const float* __restrict__ emo, float* __restrict__ out) {
  int i = blockIdx.x * 256 + threadIdx.x;  // B*1024
  int b = i >> 10, e = i & 1023;
  const float* p = emo + (long)b * LT * EMO_ + e;
  float s = 0.f;
  for (int l = 0; l < LT; ++l) s += p[(long)l * EMO_];
  out[i] = s * (1.0f / LT);
}

// per batch: A_pool[k]=emo_mean.agg[k]; route=A_pool@agg; route_dot=route.eleW2
__global__ __launch_bounds__(256)
void apool_route_kernel(const float* __restrict__ emo_mean,
                        const float* __restrict__ agg,
                        const float* __restrict__ eleW,
                        float* __restrict__ route_dot) {
  __shared__ float sm[1024];
  __shared__ float apool[128];
  __shared__ float red[256];
  const int b = blockIdx.x, tid = threadIdx.x;
  for (int e = tid; e < 1024; e += 256) sm[e] = emo_mean[b * 1024 + e];
  __syncthreads();
  if (tid < 128) {
    const float* ar = agg + (long)b * 128 * 1024 + (long)tid * 1024;
    float s = 0.f;
    for (int e = 0; e < 1024; ++e) s += sm[e] * ar[e];
    apool[tid] = s;
  }
  __syncthreads();
  float part = 0.f;
  for (int e = tid; e < 1024; e += 256) {
    const float* ac = agg + (long)b * 128 * 1024 + e;
    float r = 0.f;
    for (int k = 0; k < 128; ++k) r += apool[k] * ac[(long)k * 1024];
    part += r * eleW[1024 + e];
  }
  red[tid] = part;
  __syncthreads();
  for (int o = 128; o; o >>= 1) { if (tid < o) red[tid] += red[tid + o]; __syncthreads(); }
  if (tid == 0) route_dot[b] = red[0];
}

// R_e = tanh(emo.W1 + route_dot + b); route_emo fp16 for the Q GEMM
__global__ __launch_bounds__(128)
void route_emo_kernel(const float* __restrict__ emo, const float* __restrict__ eleW,
                      const float* __restrict__ eleB, const float* __restrict__ route_dot,
                      float* __restrict__ Re, _Float16* __restrict__ remo) {
  int row = blockIdx.x * 4 + (threadIdx.x >> 5);  // B*LT
  int lane = threadIdx.x & 31;
  int b = row >> 10;
  const float* er = emo + (long)row * EMO_;
  float acc = 0.f;
#pragma unroll 8
  for (int j = 0; j < 32; ++j) { int e = lane + 32 * j; acc += er[e] * eleW[e]; }
  acc = wave_sum(acc);
  float r = tanhf(acc + route_dot[b] + eleB[0]);
  if (lane == 0) Re[row] = r;
  _Float16* o = remo + (long)row * EMO_;
#pragma unroll 8
  for (int j = 0; j < 32; ++j) { int e = lane + 32 * j; o[e] = (_Float16)(r * er[e]); }
}

// V = agg@wv + b [B,128,28], Rp = agg@wr + b [B,128,3] (one wave per (b,k))
__global__ __launch_bounds__(128)
void vrp_kernel(const float* __restrict__ agg, const float* __restrict__ wvW,
                const float* __restrict__ wvB, const float* __restrict__ wrW,
                const float* __restrict__ wrB, float* __restrict__ V,
                float* __restrict__ Rp) {
  int idx = blockIdx.x * 4 + (threadIdx.x >> 5);  // B*128
  int lane = threadIdx.x & 31;
  const float* ar = agg + (long)idx * 1024;
  float acc = 0.f;
  if (lane < 28) {
    for (int e = 0; e < 1024; ++e) acc += ar[e] * wvW[e * 28 + lane];
    V[(long)idx * 28 + lane] = acc + wvB[lane];
  } else if (lane < 31) {
    int j = lane - 28;
    for (int e = 0; e < 1024; ++e) acc += ar[e] * wrW[e * 3 + j];
    Rp[(long)idx * 3 + j] = acc + wrB[j];
  }
}

__global__ void zero_kernel(float* __restrict__ p, int n) {
  int i = blockIdx.x * 256 + threadIdx.x;
  if (i < n) p[i] = 0.f;
}

// O = tanh(P@V/3) [B,3,Lt,28]; Rdiag[b,i] += (P@Rp[:,i]) (gate diag only)
__global__ __launch_bounds__(128)
void ov_kernel(const _Float16* __restrict__ P, const float* __restrict__ V,
               const float* __restrict__ Rp, float* __restrict__ Otanh,
               float* __restrict__ Rdiag) {
  long row = (long)blockIdx.x * 4 + (threadIdx.x >> 5);  // B*3*LT
  int lane = threadIdx.x & 31;
  int bi = (int)(row >> 10);
  int b = bi / 3, i = bi - 3 * b;
  const _Float16* pr = P + row * 128;
  float acc = 0.f;
  if (lane < 28) {
    const float* vc = V + (long)b * 128 * 28 + lane;
    for (int k = 0; k < 128; ++k) acc += (float)pr[k] * vc[(long)k * 28];
    Otanh[row * 28 + lane] = tanhf(acc * (1.0f / 3.0f));
  } else if (lane == 31) {
    const float* rc = Rp + (long)b * 128 * 3 + i;
    for (int k = 0; k < 128; ++k) acc += (float)pr[k] * rc[(long)k * 3];
    atomicAdd(&Rdiag[bi], acc);
  }
}

__global__ void gate_kernel(const float* __restrict__ Rdiag, float* __restrict__ gate) {
  int b = threadIdx.x;
  if (b >= B_) return;
  float r0 = Rdiag[b * 3 + 0] * (1.0f / LT);
  float r1 = Rdiag[b * 3 + 1] * (1.0f / LT);
  float r2 = Rdiag[b * 3 + 2] * (1.0f / LT);
  float mx = fmaxf(r0, fmaxf(r1, r2));
  float e0 = __expf(r0 - mx), e1 = __expf(r1 - mx), e2 = __expf(r2 - mx);
  float inv = 1.0f / (e0 + e1 + e2);
  gate[b * 3 + 0] = e0 * inv; gate[b * 3 + 1] = e1 * inv; gate[b * 3 + 2] = e2 * inv;
}

// out = emo * (1 + Re * sum_i gate_i * O_i[head_i(e)])
__global__ __launch_bounds__(256)
void final_kernel(const float* __restrict__ emo, const float* __restrict__ Re,
                  const float* __restrict__ Otanh, const float* __restrict__ gate,
                  float* __restrict__ out) {
  __shared__ float so[84];
  int row = blockIdx.x;           // b*1024 + l
  int b = row >> 10, l = row & 1023;
  int tid = threadIdx.x;
  if (tid < 84) {
    int i = tid / 28, h = tid - 28 * i;
    so[tid] = gate[b * 3 + i] * Otanh[((long)(b * 3 + i) * LT + l) * 28 + h];
  }
  __syncthreads();
  float r = Re[row];
  const float* er = emo + (long)row * 1024;
  float* orow = out + (long)row * 1024;
#pragma unroll
  for (int j = 0; j < 4; ++j) {
    int e = tid + 256 * j;
    float g = so[e >> 8] + so[28 + 4 + (e >> 7)] + so[56 + 12 + (e >> 6)];
    orow[e] = er[e] * (1.0f + r * g);
  }
}

// ---------------------------------------------------------------------------
extern "C" void kernel_launch(void* const* d_in, const int* in_sizes, int n_in,
                              void* d_out, int out_size, void* d_ws, size_t ws_size,
                              hipStream_t stream) {
  (void)in_sizes; (void)n_in; (void)out_size; (void)ws_size;
  const float* emo    = (const float*)d_in[0];
  const float* phr    = (const float*)d_in[1];
  const float* video  = (const float*)d_in[2];
  const float* w_v_W  = (const float*)d_in[3];
  const float* w_v_b  = (const float*)d_in[4];
  const float* agg_aW = (const float*)d_in[5];
  const float* agg_ab = (const float*)d_in[6];
  const float* agg_cW = (const float*)d_in[7];
  const float* agg_cb = (const float*)d_in[8];
  const float* ele_W  = (const float*)d_in[9];
  const float* ele_b  = (const float*)d_in[10];
  const float* wq_W   = (const float*)d_in[11];
  const float* wq_b   = (const float*)d_in[12];
  const float* wk_W   = (const float*)d_in[13];
  const float* wk_b   = (const float*)d_in[14];
  const float* wv_W   = (const float*)d_in[15];
  const float* wv_b   = (const float*)d_in[16];
  const float* wr_W   = (const float*)d_in[17];
  const float* wr_b   = (const float*)d_in[18];
  float* out = (float*)d_out;

  char* ws = (char*)d_ws;
  size_t off = 0;
  auto alloc = [&](size_t bytes) -> void* {
    void* p = ws + off;
    off = (off + bytes + 255) & ~(size_t)255;
    return p;
  };

  // transposed fp16 weights ([N,K] row-major -> async-DMA B staging)
  _Float16* wvT    = (_Float16*)alloc((size_t)1024 * 1024 * 2);
  _Float16* agg_aT = (_Float16*)alloc((size_t)128 * 1024 * 2);
  _Float16* agg_cT = (_Float16*)alloc((size_t)1024 * 1024 * 2);
  _Float16* wqT    = (_Float16*)alloc((size_t)448 * 1024 * 2);
  _Float16* wkT    = (_Float16*)alloc((size_t)448 * 1024 * 2);
  _Float16* phrp_h  = (_Float16*)alloc((size_t)B_ * 256 * 1024 * 2);
  _Float16* comb_h  = (_Float16*)alloc((size_t)B_ * 512 * 1024 * 2);
  float*    logits  = (float*)   alloc((size_t)B_ * 512 * 128 * 4);
  _Float16* attnT_h = (_Float16*)alloc((size_t)B_ * 128 * 512 * 2);
  _Float16* CE_h    = (_Float16*)alloc((size_t)B_ * 512 * 1024 * 2);
  float*    agg_f   = (float*)   alloc((size_t)B_ * 128 * 1024 * 4);
  _Float16* agg_h   = (_Float16*)alloc((size_t)B_ * 128 * 1024 * 2);
  float*    emean   = (float*)   alloc((size_t)B_ * 1024 * 4);
  float*    rdot    = (float*)   alloc((size_t)B_ * 4);
  float*    Re      = (float*)   alloc((size_t)B_ * LT * 4);
  _Float16* remo_h  = (_Float16*)alloc((size_t)B_ * LT * 1024 * 2);
  _Float16* Q_h     = (_Float16*)alloc((size_t)B_ * LT * 448 * 2);
  _Float16* Kp_h    = (_Float16*)alloc((size_t)B_ * 128 * 448 * 2);
  float*    S_f     = (float*)   alloc((size_t)B_ * 3 * LT * 128 * 4);
  _Float16* P_h     = (_Float16*)alloc((size_t)B_ * 3 * LT * 128 * 2);
  float*    V_f     = (float*)   alloc((size_t)B_ * 128 * 28 * 4);
  float*    Rp_f    = (float*)   alloc((size_t)B_ * 128 * 3 * 4);
  float*    Ot_f    = (float*)   alloc((size_t)B_ * 3 * LT * 28 * 4);
  float*    Rdiag   = (float*)   alloc((size_t)B_ * 3 * 4);
  float*    gate    = (float*)   alloc((size_t)B_ * 3 * 4);

  auto gemm = [&](const _Float16* A, long sA, int lda, const _Float16* Bm, long sB,
                  int ldb, int bT, float* C32, long sC32, _Float16* C16, long sC16,
                  int ldc, const float* bias, float alpha, int M, int Nn, int Kd) {
    dim3 grid(M / 64, Nn / 64, B_);
    wmma_gemm_kernel<<<grid, 128, 0, stream>>>(A, sA, lda, Bm, sB, ldb, bT, C32,
                                               sC32, C16, sC16, ldc, bias, alpha, Kd);
  };
  auto transpose_w = [&](const float* w, _Float16* wT, int K, int N) {
    transpose_w_kernel<<<(K * N + 255) / 256, 256, 0, stream>>>(w, wT, K, N);
  };

  // one-time weight conversion+transpose
  transpose_w(w_v_W, wvT, 1024, 1024);
  transpose_w(agg_aW, agg_aT, 1024, 128);
  transpose_w(agg_cW, agg_cT, 1024, 1024);
  transpose_w(wq_W, wqT, 1024, 448);
  transpose_w(wk_W, wkT, 1024, 448);

  // feature_aggregation
  pool_phr_kernel<<<B_ * 256 * 1024 / 256, 256, 0, stream>>>(phr, phrp_h);
  video_to_combine_kernel<<<B_ * 256 * 1024 / 256, 256, 0, stream>>>(video, comb_h);
  gemm(phrp_h, 256L * 1024, 1024, wvT, 0, 1024, 1, nullptr, 0, comb_h, 512L * 1024,
       1024, w_v_b, 1.0f, 256, 1024, 1024);                       // combine[:,0:256]
  gemm(comb_h, 512L * 1024, 1024, agg_aT, 0, 1024, 1, logits, 512L * 128, nullptr, 0,
       128, agg_ab, 1.0f, 512, 128, 1024);                        // attn logits
  softmax128_T_kernel<<<B_ * 512 / 4, 128, 0, stream>>>(logits, attnT_h);
  gemm(comb_h, 512L * 1024, 1024, agg_cT, 0, 1024, 1, nullptr, 0, CE_h, 512L * 1024,
       1024, agg_cb, 1.0f, 512, 1024, 1024);                      // CE
  gemm(attnT_h, 128L * 512, 512, CE_h, 512L * 1024, 1024, 0, agg_f, 128L * 1024,
       agg_h, 128L * 1024, 1024, nullptr, 1.0f, 128, 1024, 512);  // agg_feat

  // element_routing (pooled-first A_pool; scalar route projection)
  emo_mean_kernel<<<B_ * 1024 / 256, 256, 0, stream>>>(emo, emean);
  apool_route_kernel<<<B_, 256, 0, stream>>>(emean, agg_f, ele_W, rdot);
  route_emo_kernel<<<B_ * LT / 4, 128, 0, stream>>>(emo, ele_W, ele_b, rdot, Re, remo_h);

  // subspace_routing projections
  gemm(remo_h, (long)LT * 1024, 1024, wqT, 0, 1024, 1, nullptr, 0, Q_h, (long)LT * 448,
       448, wq_b, 1.0f, LT, 448, 1024);                           // Q
  gemm(agg_h, 128L * 1024, 1024, wkT, 0, 1024, 1, nullptr, 0, Kp_h, 128L * 448,
       448, wk_b, 1.0f, 128, 448, 1024);                          // Kp
  vrp_kernel<<<B_ * 128 / 4, 128, 0, stream>>>(agg_f, wv_W, wv_b, wr_W, wr_b, V_f, Rp_f);

  // per-subspace scores: S_i = Q[:,blk_i] @ Kp[:,blk_i]^T / sqrt(448)
  const float rTemp = 0.047245559f;  // 1/sqrt(448)
  const int dOff[3] = {0, 256, 384};
  const int dLen[3] = {256, 128, 64};
  for (int i = 0; i < 3; ++i) {
    gemm(Q_h + dOff[i], (long)LT * 448, 448, Kp_h + dOff[i], 128L * 448, 448, 1,
         S_f + (long)i * LT * 128, 3L * LT * 128, nullptr, 0, 128, nullptr, rTemp,
         LT, 128, dLen[i]);
  }
  softmax128_kernel<<<B_ * 3 * LT / 4, 128, 0, stream>>>(S_f, P_h);

  zero_kernel<<<1, 256, 0, stream>>>(Rdiag, B_ * 3);
  ov_kernel<<<B_ * 3 * LT / 4, 128, 0, stream>>>(P_h, V_f, Rp_f, Ot_f, Rdiag);
  gate_kernel<<<1, 32, 0, stream>>>(Rdiag, gate);
  final_kernel<<<B_ * LT, 256, 0, stream>>>(emo, Re, Ot_f, gate, out);
}
